// HyperFLINTNet_49100066127964
// MI455X (gfx1250) — matrix operand order repaired and checked by reading, not compiled, not run on hardware
//
#include <hip/hip_runtime.h>

// ---------------------------------------------------------------------------
// HyperFLINT forward for MI455X (gfx1250, wave32, WMMA).
//  - hypernet MLP/conv1d: tiny, single block.
//  - fc layer (HBM-bound, 180MB of fc_w): coalesced float4 VALU kernel.
//  - weights for all 64-out conv layers pre-packed to bf16 [ko][m][ic] so the
//    WMMA kernel's A-tile is one contiguous block (b128 loads, no conversion).
//  - main conv3d layers: implicit-GEMM on v_wmma_f32_16x16x32_bf16,
//    channels-last bf16 activations, fp32 accumulation, LDS-staged tiles,
//    2 WMMAs per barrier pair (full K=64 slice staged per kernel offset).
//  - c00 (Cin=3) and the three transposed convs: direct VALU kernels.
// ---------------------------------------------------------------------------

typedef __attribute__((ext_vector_type(16))) __bf16 v16bf;
typedef __attribute__((ext_vector_type(8)))  float  v8f;

static __device__ __forceinline__ unsigned short f2bfbits(float f) {
  unsigned u = __builtin_bit_cast(unsigned, f);
  unsigned r = (u + 0x7FFFu + ((u >> 16) & 1u)) >> 16;   // RNE
  return (unsigned short)r;
}
static __device__ __forceinline__ float bfbits2f(unsigned short u) {
  return __builtin_bit_cast(float, ((unsigned)u) << 16);
}
static __device__ __forceinline__ __bf16 us2bf(unsigned short u) {
  return __builtin_bit_cast(__bf16, u);
}
static __device__ __forceinline__ float preluf(float v, float a) {
  return v >= 0.f ? v : a * v;
}

// ---------------------------------------------------------------------------
// Hypernet MLP (3->16->32->3) + conv1d (1->16->32, len 3)  ->  h [4][96]
// ---------------------------------------------------------------------------
__global__ void hyper_mlp_cnn_kernel(
    const float* __restrict__ params,
    const float* __restrict__ w1, const float* __restrict__ b1, const float* __restrict__ a1,
    const float* __restrict__ w2, const float* __restrict__ b2, const float* __restrict__ a2,
    const float* __restrict__ w3, const float* __restrict__ b3, const float* __restrict__ a3,
    const float* __restrict__ cw1, const float* __restrict__ cbias1, const float* __restrict__ ca1,
    const float* __restrict__ cw2, const float* __restrict__ cbias2, const float* __restrict__ ca2,
    float* __restrict__ hout) {
  __shared__ float s1[16], s2[32], s3[3], sc1[16][3];
  int t = threadIdx.x;
  for (int b = 0; b < 4; ++b) {
    if (t < 16) {
      float acc = b1[t];
      for (int i = 0; i < 3; ++i) acc += params[b * 3 + i] * w1[t * 3 + i];
      s1[t] = preluf(acc, *a1);
    }
    __syncthreads();
    if (t < 32) {
      float acc = b2[t];
      for (int i = 0; i < 16; ++i) acc += s1[i] * w2[t * 16 + i];
      s2[t] = preluf(acc, *a2);
    }
    __syncthreads();
    if (t < 3) {
      float acc = b3[t];
      for (int i = 0; i < 32; ++i) acc += s2[i] * w3[t * 32 + i];
      s3[t] = preluf(acc, *a3);
    }
    __syncthreads();
    if (t < 48) {  // conv1d 1->16, k=3, pad 1
      int o = t / 3, p = t % 3;
      float acc = cbias1[o];
      for (int k = 0; k < 3; ++k) {
        int ip = p + k - 1;
        if (ip >= 0 && ip < 3) acc += s3[ip] * cw1[o * 3 + k];
      }
      sc1[o][p] = preluf(acc, *ca1);
    }
    __syncthreads();
    if (t < 96) {  // conv1d 16->32, k=3, pad 1; reshape (c*3 + t)
      int o = t / 3, p = t % 3;
      float acc = cbias2[o];
      for (int i = 0; i < 16; ++i)
        for (int k = 0; k < 3; ++k) {
          int ip = p + k - 1;
          if (ip >= 0 && ip < 3) acc += sc1[i][ip] * cw2[(o * 16 + i) * 3 + k];
        }
      hout[b * 96 + o * 3 + p] = preluf(acc, *ca2);
    }
    __syncthreads();
  }
}

// ---------------------------------------------------------------------------
// wflat[b][t] = h[b][:] . fc_w[t][:] + fc_b[t]     (HBM-bound on fc_w)
// ---------------------------------------------------------------------------
__global__ __launch_bounds__(256) void hyper_fc_kernel(
    const float* __restrict__ h, const float* __restrict__ fw,
    const float* __restrict__ fb, float* __restrict__ wflat, int total) {
  __shared__ float sh[4][96];
  int t = threadIdx.x;
  for (int i = t; i < 4 * 96; i += 256) (&sh[0][0])[i] = h[i];
  __syncthreads();
  int row = blockIdx.x * 256 + t;
  if (row >= total) return;
  const float4* r = (const float4*)(fw + (long)row * 96);
  float acc0 = 0.f, acc1 = 0.f, acc2 = 0.f, acc3 = 0.f;
  #pragma unroll 6
  for (int j = 0; j < 24; ++j) {
    float4 w4 = r[j];
    int c = j * 4;
    acc0 += w4.x * sh[0][c] + w4.y * sh[0][c + 1] + w4.z * sh[0][c + 2] + w4.w * sh[0][c + 3];
    acc1 += w4.x * sh[1][c] + w4.y * sh[1][c + 1] + w4.z * sh[1][c + 2] + w4.w * sh[1][c + 3];
    acc2 += w4.x * sh[2][c] + w4.y * sh[2][c + 1] + w4.z * sh[2][c + 2] + w4.w * sh[2][c + 3];
    acc3 += w4.x * sh[3][c] + w4.y * sh[3][c + 1] + w4.z * sh[3][c + 2] + w4.w * sh[3][c + 3];
  }
  float bb = fb[row];
  wflat[row]             = acc0 + bb;
  wflat[total + row]     = acc1 + bb;
  wflat[2 * total + row] = acc2 + bb;
  wflat[3 * total + row] = acc3 + bb;
}

// ---------------------------------------------------------------------------
// Pack conv weights fp32 OIDHW -> bf16 [ko][m][ic] (contiguous A-tiles).
// ---------------------------------------------------------------------------
__global__ __launch_bounds__(256) void pack_weights_kernel(
    const float* __restrict__ w, long wStride,
    unsigned short* __restrict__ wp, long pStride, int Cin, int KVOL) {
  int b = blockIdx.y;
  const float* wb = w + (long)b * wStride;
  unsigned short* pb = wp + (long)b * pStride;
  int n = 64 * Cin * KVOL;
  for (int i = blockIdx.x * 256 + threadIdx.x; i < n; i += gridDim.x * 256) {
    int ko = i / (64 * Cin);
    int rem = i % (64 * Cin);
    int m = rem / Cin;
    int ic = rem % Cin;
    pb[i] = f2bfbits(wb[(m * Cin + ic) * KVOL + ko]);
  }
}

// ---------------------------------------------------------------------------
// c00: conv3d 3->32, k=4, s=2, p=2, 64^3 -> 33^3. Direct VALU (Cin=3).
// ---------------------------------------------------------------------------
__global__ __launch_bounds__(256) void c00_direct_kernel(
    const float* __restrict__ x, const float* __restrict__ wflat,
    unsigned short* __restrict__ out, long total) {
  const int DO = 33, DI = 64;
  int idx = blockIdx.x * 256 + threadIdx.x;
  if (idx >= DO * DO * DO * 32) return;
  int b = blockIdx.y;
  int oc = idx & 31;
  int s = idx >> 5;
  int oz = s / (DO * DO); int r = s % (DO * DO); int oy = r / DO; int ox = r % DO;
  const float* wf = wflat + (long)b * total;
  float acc = wf[6144 + oc];                      // c00_b
  const float* xb = x + (long)b * 3 * DI * DI * DI;
  for (int kd = 0; kd < 4; ++kd) {
    int iz = oz * 2 - 2 + kd; if ((unsigned)iz >= (unsigned)DI) continue;
    for (int kh = 0; kh < 4; ++kh) {
      int iy = oy * 2 - 2 + kh; if ((unsigned)iy >= (unsigned)DI) continue;
      for (int kw = 0; kw < 4; ++kw) {
        int ix = ox * 2 - 2 + kw; if ((unsigned)ix >= (unsigned)DI) continue;
        long sp = ((long)iz * DI + iy) * DI + ix;
        int ko = (kd * 4 + kh) * 4 + kw;
        #pragma unroll
        for (int ic = 0; ic < 3; ++ic)
          acc += xb[(long)ic * (DI * DI * DI) + sp] * wf[(oc * 3 + ic) * 64 + ko];
      }
    }
  }
  acc = preluf(acc, 0.25f);
  out[((long)b * (DO * DO * DO) + s) * 32 + oc] = f2bfbits(acc);
}

// ---------------------------------------------------------------------------
// Implicit-GEMM conv3d on WMMA bf16. Cout=64, CIN/K3 compile-time.
// Block = 8 waves = 4(M) x 2(N) => 64x32 tile. Per kernel offset, the whole
// K=CIN slice is staged (As 64xCIN, Bs 32xCIN) -> CIN/32 WMMAs per barrier.
// Weights pre-packed bf16 [ko][m][ic]. Activations channels-last bf16.
// ---------------------------------------------------------------------------
template <int CIN, int K3>
__global__ __launch_bounds__(256) void conv3d_wmma_kernel(
    const unsigned short* __restrict__ inb,
    const unsigned short* __restrict__ wpb, long wStride,
    const float* __restrict__ bb, long bStride,
    const float* __restrict__ alphaPtr, float alphaConst, float outScale,
    unsigned short* __restrict__ outb,
    int stride, int pad, int Din, int Dout) {
  constexpr int KVOL = K3 * K3 * K3;
  __shared__ unsigned short As[64 * CIN];   // [m][ic] slice for one ko
  __shared__ unsigned short Bs[32 * CIN];   // [n][ic]
  const int tid = threadIdx.x;
  const int b = blockIdx.y;
  const int Ns = Dout * Dout * Dout;
  const unsigned short* in = inb + (long)b * Din * Din * Din * CIN;
  const unsigned short* wp = wpb + (long)b * wStride;
  const float* bias = bb + (long)b * bStride;
  unsigned short* out = outb + (long)b * Ns * 64;
  const int tileBase = blockIdx.x * 32;

  // B-loader role: thread -> (n within tile, contiguous channel group)
  constexpr int PB = CIN / 8;               // ushorts per thread (8 or 4)
  const int bn = tid >> 3;
  const int bg = (tid & 7) * PB;
  int ng = tileBase + bn;
  bool nvalid = ng < Ns;
  int oz = 0, oy = 0, ox = 0;
  if (nvalid) { oz = ng / (Dout * Dout); int rr = ng % (Dout * Dout); oy = rr / Dout; ox = rr % Dout; }
  // A-loader role: contiguous block copy, PER ushorts per thread
  constexpr int PER = (64 * CIN) / 256;     // 16 or 8 ushorts

  // Wave / lane geometry
  const int lane = tid & 31;
  const int wv = tid >> 5;
  const int wm = wv & 3;       // M tile (0..3)
  const int wn = wv >> 2;      // N tile (0..1)
  const int hi = (lane >= 16) ? 1 : 0;
  const int l15 = lane & 15;

  v8f acc = {};
  for (int ko = 0; ko < KVOL; ++ko) {
    int kd = ko / (K3 * K3); int kr = ko % (K3 * K3); int kh = kr / K3; int kw = kr % K3;
    int iz = oz * stride - pad + kd;
    int iy = oy * stride - pad + kh;
    int ix = ox * stride - pad + kw;
    bool pvalid = nvalid && (unsigned)iz < (unsigned)Din &&
                  (unsigned)iy < (unsigned)Din && (unsigned)ix < (unsigned)Din;
    long spos = pvalid ? (((long)iz * Din + iy) * Din + ix) : 0;
    __syncthreads();
    { // A slice: contiguous bf16 block [64][CIN] at wp + ko*64*CIN
      const uint4* src = (const uint4*)(wp + (long)ko * 64 * CIN + tid * PER);
      uint4* dst = (uint4*)(As + tid * PER);
      #pragma unroll
      for (int i = 0; i < PER / 8; ++i) dst[i] = src[i];
    }
    { // B slice: channels-last contiguous bf16 per position
      if constexpr (PB == 8) {
        uint4 v = {0, 0, 0, 0};
        if (pvalid) v = *(const uint4*)(in + spos * CIN + bg);
        *(uint4*)(Bs + bn * CIN + bg) = v;
      } else {
        uint2 v = {0, 0};
        if (pvalid) v = *(const uint2*)(in + spos * CIN + bg);
        *(uint2*)(Bs + bn * CIN + bg) = v;
      }
    }
    __syncthreads();
    #pragma unroll
    for (int cc = 0; cc < CIN / 32; ++cc) {
      v16bf af, bf;
      const unsigned short* ap = As + (wm * 16 + l15) * CIN + cc * 32;
      const int abase = hi ? 16 : 0;
      #pragma unroll
      for (int e = 0; e < 8; ++e) {        // A: contiguous k = 16*hi + e, matching B
        af[e]     = us2bf(ap[abase + e]);
        af[e + 8] = us2bf(ap[abase + 8 + e]);
      }
      const unsigned short* bp = Bs + (wn * 16 + l15) * CIN + cc * 32;
      const int bbase = hi ? 16 : 0;
      #pragma unroll
      for (int e = 0; e < 16; ++e) bf[e] = us2bf(bp[bbase + e]);
      acc = __builtin_amdgcn_wmma_f32_16x16x32_bf16(
          false, af, false, bf, (short)0, acc, false, false);
    }
  }
  // Epilogue: D layout M = wm*16 + (lane>=16?8:0) + j, N = wn*16 + lane%16
  float alpha = alphaPtr ? *alphaPtr : alphaConst;
  int N = wn * 16 + l15;
  int og = tileBase + N;
  if (og < Ns) {
    int mb = wm * 16 + (hi ? 8 : 0);
    #pragma unroll
    for (int j = 0; j < 8; ++j) {
      int M = mb + j;
      float v = acc[j] + bias[M];
      v = preluf(v, alpha);
      v *= outScale;
      out[(long)og * 64 + M] = f2bfbits(v);
    }
  }
}

// ---------------------------------------------------------------------------
// Transposed conv3d (k=4, stride=2, pad=1). Direct gather: <=2 taps per dim.
// w: (Cin, Cout, 4,4,4) fp32. mode 0: bf16 channels-last out.
// mode 1: fp32 flows/masks split into d_out.
// ---------------------------------------------------------------------------
__global__ __launch_bounds__(256) void deconv3d_kernel(
    const unsigned short* __restrict__ inb, const float* __restrict__ w,
    const float* __restrict__ bias, unsigned short* __restrict__ outb,
    float* __restrict__ outf, int Cin, int Cout, int Din, int Dout, int mode) {
  int b = blockIdx.y;
  long nOut = (long)Dout * Dout * Dout * Cout;
  long idx = (long)blockIdx.x * 256 + threadIdx.x;
  if (idx >= nOut) return;
  int oc = (int)(idx % Cout);
  long s = idx / Cout;
  int oz = (int)(s / ((long)Dout * Dout));
  int r = (int)(s % ((long)Dout * Dout));
  int oy = r / Dout, ox = r % Dout;
  const unsigned short* in = inb + (long)b * Din * Din * Din * Cin;
  int izs[2], wzs[2], nz = 0, iys[2], wys[2], nyc = 0, ixs[2], wxs[2], nxc = 0;
  for (int k = 0; k < 4; ++k) {
    int j = oz + k - 2;
    if (j >= 0 && (j & 1) == 0 && (j >> 1) < Din) { izs[nz] = j >> 1; wzs[nz] = 3 - k; ++nz; }
    j = oy + k - 2;
    if (j >= 0 && (j & 1) == 0 && (j >> 1) < Din) { iys[nyc] = j >> 1; wys[nyc] = 3 - k; ++nyc; }
    j = ox + k - 2;
    if (j >= 0 && (j & 1) == 0 && (j >> 1) < Din) { ixs[nxc] = j >> 1; wxs[nxc] = 3 - k; ++nxc; }
  }
  float acc = bias[oc];
  for (int a = 0; a < nz; ++a)
    for (int c = 0; c < nyc; ++c)
      for (int d = 0; d < nxc; ++d) {
        long sp = (((long)izs[a] * Din + iys[c]) * Din + ixs[d]) * Cin;
        int koff = (wzs[a] * 4 + wys[c]) * 4 + wxs[d];
        const unsigned short* ip = in + sp;
        const float* wp = w + (long)oc * 64 + koff;   // + ic*Cout*64
        float partial = 0.f;
        for (int ic = 0; ic < Cin; ++ic)
          partial += bfbits2f(ip[ic]) * wp[(long)ic * Cout * 64];
        acc += partial;
      }
  if (mode == 0) {
    outb[((long)b * Dout * Dout * Dout + s) * Cout + oc] = f2bfbits(acc);
  } else {  // split: flows = ch 0..5, masks = ch 6 (concatenated in d_out)
    long V = (long)Dout * Dout * Dout;
    if (oc < 6) outf[((long)b * 6 + oc) * V + s] = acc;
    else        outf[(long)4 * 6 * V + (long)b * V + s] = acc;
  }
}

// ---------------------------------------------------------------------------
extern "C" void kernel_launch(void* const* d_in, const int* in_sizes, int n_in,
                              void* d_out, int out_size, void* d_ws, size_t ws_size,
                              hipStream_t stream) {
  (void)in_sizes; (void)n_in; (void)out_size; (void)ws_size;
  const float* x       = (const float*)d_in[0];
  const float* params  = (const float*)d_in[1];
  const float* mlp_w1  = (const float*)d_in[2];
  const float* mlp_b1  = (const float*)d_in[3];
  const float* mlp_a1  = (const float*)d_in[4];
  const float* mlp_w2  = (const float*)d_in[5];
  const float* mlp_b2  = (const float*)d_in[6];
  const float* mlp_a2  = (const float*)d_in[7];
  const float* mlp_w3  = (const float*)d_in[8];
  const float* mlp_b3  = (const float*)d_in[9];
  const float* mlp_a3  = (const float*)d_in[10];
  const float* cnn_w1  = (const float*)d_in[11];
  const float* cnn_b1  = (const float*)d_in[12];
  const float* cnn_a1  = (const float*)d_in[13];
  const float* cnn_w2  = (const float*)d_in[14];
  const float* cnn_b2  = (const float*)d_in[15];
  const float* cnn_a2  = (const float*)d_in[16];
  const float* fc_w    = (const float*)d_in[17];
  const float* fc_b    = (const float*)d_in[18];
  const float* prelu_a = (const float*)d_in[19];
  const float* cbw[7]; const float* cbb[7];
  for (int j = 0; j < 7; ++j) { cbw[j] = (const float*)d_in[20 + 2 * j]; cbb[j] = (const float*)d_in[21 + 2 * j]; }
  const float* dec0_w = (const float*)d_in[34];
  const float* dec0_b = (const float*)d_in[35];
  const float* dec1_w = (const float*)d_in[36];
  const float* dec1_b = (const float*)d_in[37];
  const float* dec2_w = (const float*)d_in[38];
  const float* dec2_b = (const float*)d_in[39];
  float* outp = (float*)d_out;

  const long TOTAL = 469280;
  char* wsb = (char*)d_ws;
  float* hbuf  = (float*)(wsb + 0);                        // 4x96 fp32
  float* wflat = (float*)(wsb + 4096);                     // 4xTOTAL fp32
  unsigned short* c01p = (unsigned short*)(wsb + 7512576); // 4x131072 bf16
  unsigned short* cb0p = (unsigned short*)(wsb + 8561152); // 4x110592 bf16
  unsigned short* cb1p = (unsigned short*)(wsb + 9445888);
  unsigned short* cb2p = (unsigned short*)(wsb + 10330624);
  unsigned short* cbsp[7];                                 // shared cb3..cb9 packed
  for (int j = 0; j < 7; ++j) cbsp[j] = (unsigned short*)(wsb + 11215360 + (long)j * 221184);
  unsigned short* actA = (unsigned short*)(wsb + 12763648); // 16MB bf16 ping
  unsigned short* actB = (unsigned short*)(wsb + 29540864); // 16MB bf16 pong

  // wflat offsets: c00_w 0, c00_b 6144, c01_w 6176, c01_b 137248,
  // cb0_w 137312, cb0_b 247904, cb1_w 247968, cb1_b 358560,
  // cb2_w 358624, cb2_b 469216.

  hyper_mlp_cnn_kernel<<<1, 96, 0, stream>>>(
      params, mlp_w1, mlp_b1, mlp_a1, mlp_w2, mlp_b2, mlp_a2,
      mlp_w3, mlp_b3, mlp_a3, cnn_w1, cnn_b1, cnn_a1, cnn_w2, cnn_b2, cnn_a2, hbuf);
  hyper_fc_kernel<<<(int)((TOTAL + 255) / 256), 256, 0, stream>>>(
      hbuf, fc_w, fc_b, wflat, (int)TOTAL);

  // Pack weights to bf16 [ko][m][ic]
  pack_weights_kernel<<<dim3(128, 4), 256, 0, stream>>>(wflat + 6176,   TOTAL, c01p, 131072, 32, 64);
  pack_weights_kernel<<<dim3(128, 4), 256, 0, stream>>>(wflat + 137312, TOTAL, cb0p, 110592, 64, 27);
  pack_weights_kernel<<<dim3(128, 4), 256, 0, stream>>>(wflat + 247968, TOTAL, cb1p, 110592, 64, 27);
  pack_weights_kernel<<<dim3(128, 4), 256, 0, stream>>>(wflat + 358624, TOTAL, cb2p, 110592, 64, 27);
  for (int j = 0; j < 7; ++j)
    pack_weights_kernel<<<dim3(128, 1), 256, 0, stream>>>(cbw[j], 0, cbsp[j], 0, 64, 27);

  c00_direct_kernel<<<dim3((33 * 33 * 33 * 32 + 255) / 256, 4), 256, 0, stream>>>(
      x, wflat, actA, TOTAL);

  auto tiles = [](int Dout) { return (Dout * Dout * Dout + 31) / 32; };

  // c01: 32->64, k4 s2 p2, 33^3 -> 17^3 (dynamic, PReLU 0.25)
  conv3d_wmma_kernel<32, 4><<<dim3(tiles(17), 4), 256, 0, stream>>>(
      actA, c01p, 131072, wflat + 137248, TOTAL, nullptr, 0.25f, 1.f, actB, 2, 2, 33, 17);
  // cb0..cb2: 64->64, k3 s1 p1 @ 17^3 (dynamic)
  conv3d_wmma_kernel<64, 3><<<dim3(tiles(17), 4), 256, 0, stream>>>(
      actB, cb0p, 110592, wflat + 247904, TOTAL, nullptr, 0.25f, 1.f, actA, 1, 1, 17, 17);
  conv3d_wmma_kernel<64, 3><<<dim3(tiles(17), 4), 256, 0, stream>>>(
      actA, cb1p, 110592, wflat + 358560, TOTAL, nullptr, 0.25f, 1.f, actB, 1, 1, 17, 17);
  conv3d_wmma_kernel<64, 3><<<dim3(tiles(17), 4), 256, 0, stream>>>(
      actB, cb2p, 110592, wflat + 469216, TOTAL, nullptr, 0.25f, 1.f, actA, 1, 1, 17, 17);
  // cb3: k3 s2 p1, 17^3 -> 9^3 (theta, PReLU prelu_a)
  conv3d_wmma_kernel<64, 3><<<dim3(tiles(9), 4), 256, 0, stream>>>(
      actA, cbsp[0], 0, cbb[0], 0, prelu_a, 0.f, 1.f, actB, 2, 1, 17, 9);
  // cb4..cb6 @ 9^3
  conv3d_wmma_kernel<64, 3><<<dim3(tiles(9), 4), 256, 0, stream>>>(
      actB, cbsp[1], 0, cbb[1], 0, prelu_a, 0.f, 1.f, actA, 1, 1, 9, 9);
  conv3d_wmma_kernel<64, 3><<<dim3(tiles(9), 4), 256, 0, stream>>>(
      actA, cbsp[2], 0, cbb[2], 0, prelu_a, 0.f, 1.f, actB, 1, 1, 9, 9);
  conv3d_wmma_kernel<64, 3><<<dim3(tiles(9), 4), 256, 0, stream>>>(
      actB, cbsp[3], 0, cbb[3], 0, prelu_a, 0.f, 1.f, actA, 1, 1, 9, 9);
  // dec0: 64->64 deconv, 9^3 -> 18^3 (no activation)
  deconv3d_kernel<<<dim3((int)(((long)18 * 18 * 18 * 64 + 255) / 256), 4), 256, 0, stream>>>(
      actA, dec0_w, dec0_b, actB, nullptr, 64, 64, 9, 18, 0);
  // cb7..cb9 @ 18^3; fold h = h + h into cb9 outScale=2
  conv3d_wmma_kernel<64, 3><<<dim3(tiles(18), 4), 256, 0, stream>>>(
      actB, cbsp[4], 0, cbb[4], 0, prelu_a, 0.f, 1.f, actA, 1, 1, 18, 18);
  conv3d_wmma_kernel<64, 3><<<dim3(tiles(18), 4), 256, 0, stream>>>(
      actA, cbsp[5], 0, cbb[5], 0, prelu_a, 0.f, 1.f, actB, 1, 1, 18, 18);
  conv3d_wmma_kernel<64, 3><<<dim3(tiles(18), 4), 256, 0, stream>>>(
      actB, cbsp[6], 0, cbb[6], 0, prelu_a, 0.f, 2.f, actA, 1, 1, 18, 18);
  // dec1: 64->32 deconv, 18^3 -> 36^3
  deconv3d_kernel<<<dim3((int)(((long)36 * 36 * 36 * 32 + 255) / 256), 4), 256, 0, stream>>>(
      actA, dec1_w, dec1_b, actB, nullptr, 64, 32, 18, 36, 0);
  // dec2: 32->7 deconv, 36^3 -> 72^3, split flows/masks into d_out (fp32)
  deconv3d_kernel<<<dim3((int)(((long)72 * 72 * 72 * 7 + 255) / 256), 4), 256, 0, stream>>>(
      actB, dec2_w, dec2_b, nullptr, outp, 32, 7, 36, 72, 1);
}
